// GwACLSTM_28123445854592
// MI455X (gfx1250) — matrix operand order repaired and checked by reading, not compiled, not run on hardware
//
#include <hip/hip_runtime.h>
#include <hip/hip_bf16.h>

#define NND 512
#define NE 2048
#define INF 64
#define HIDN 256
#define MSGN 128
#define OUTC 16
#define NP 4
#define GATES 1024
#define KCAT 384
#define MAXD 64
#define MAX_STEPS 5120
#define QCAP (MAX_STEPS + MAXD)

typedef __attribute__((ext_vector_type(16))) __bf16 v16bf;
typedef __attribute__((ext_vector_type(8)))  float  v8f;
typedef __attribute__((ext_vector_type(4)))  unsigned v4u;
typedef __attribute__((ext_vector_type(8)))  unsigned v8u;

union BFU { unsigned short u; __bf16 b; };

__device__ __forceinline__ unsigned short f2bf(float f) {
  unsigned u = __float_as_uint(f);
  if ((u & 0x7F800000u) == 0x7F800000u) return (unsigned short)(u >> 16); // inf/nan passthrough
  return (unsigned short)((u + 0x7FFFu + ((u >> 16) & 1u)) >> 16);        // RNE
}
__device__ __forceinline__ __bf16 bfbits(unsigned short s) { BFU t; t.u = s; return t.b; }

// ---------------- workspace layout (bytes, 256-aligned) ----------------
#define OFF_HDR   0
#define OFF_DEGS  256
#define OFF_NBRS  2304
#define OFF_QN    133376
#define OFF_QM    154112
#define OFF_HID   2808320
#define OFF_CELL  3332608
#define OFF_WCAT  3856896   // 64*12*512 bf16
#define OFF_NMP   4643328   // 8*12*512 bf16
#define OFF_ENCP  4741632   // 16*2*512 bf16
#define OFF_DECP  4774400   // 64*512 bf16
#define OFF_XABF  4839936   // 512*64 bf16
#define OFF_BSUM  4905472   // 1024 f32

// ---------------- graph prep: adjacency bitset in LDS ----------------
__global__ void __launch_bounds__(1024) prep_graph(
    const int* __restrict__ edge_index, const int* __restrict__ starts,
    const float* __restrict__ first_message,
    int* hdr, int* degs, int* nbrs, int* q_nodes, float* q_msgs) {
  __shared__ unsigned adj[NND * 16];   // 512x512 bits = 32 KB
  __shared__ int s_S0;
  int tid = threadIdx.x;
  for (int i = tid; i < NND * 16; i += blockDim.x) adj[i] = 0;
  __syncthreads();
  for (int e = tid; e < NE; e += blockDim.x) {
    int s = edge_index[e], d = edge_index[NE + e];
    atomicOr(&adj[s * 16 + (d >> 5)], 1u << (d & 31));
    atomicOr(&adj[d * 16 + (s >> 5)], 1u << (s & 31));
  }
  __syncthreads();
  if (tid < NND) {
    int cnt = 0;
    for (int w = 0; w < 16; w++) {
      unsigned m = adj[tid * 16 + w];
      while (m) {
        int b = __ffs(m) - 1;
        m &= m - 1;
        if (cnt < MAXD) nbrs[tid * MAXD + cnt] = w * 32 + b;  // sorted ascending
        cnt++;
      }
    }
    degs[tid] = cnt < MAXD ? cnt : MAXD;
  }
  if (tid == 0) {
    int c = 0;
    for (int n = 0; n < NND; n++) if (starts[n]) q_nodes[c++] = n;
    s_S0 = c;
    hdr[0] = c;
  }
  __syncthreads();
  int S0 = s_S0;
  for (int i = tid; i < S0 * MSGN; i += blockDim.x) {
    int q = i / MSGN, c = i % MSGN;
    q_msgs[(long)q * MSGN + c] = first_message[q_nodes[q] * MSGN + c];
  }
}

// ---------------- pack all weights to bf16 WMMA B-fragment layout ----------------
// B fragment (16x16x32 bf16): lane 0-15 -> col n=lane, K = ks*32 + (h<8 ? h : h+8) (h=0..15)
//                             lane 16-31 -> col n=lane-16, K = ks*32 + 8 + (h<8 ? h : h+8)
__global__ void pack_weights(
    const float* __restrict__ W_ih, const float* __restrict__ W_hh,
    const float* __restrict__ nm_W, const float* __restrict__ enc_W,
    const float* __restrict__ dec_W, const float* __restrict__ xa,
    const float* __restrict__ b_ih, const float* __restrict__ b_hh,
    unsigned short* wcat, unsigned short* nmp, unsigned short* encp,
    unsigned short* decp, unsigned short* xabf, float* bsum, float* cell) {
  long idx = (long)blockIdx.x * blockDim.x + threadIdx.x;
  if (idx < 393216) {            // Wcat = [W_ih | W_hh] : 64 otiles x 12 ksteps
    int tile = idx >> 9, w = idx & 511;
    int otile = tile / 12, ks = tile % 12;
    int lane = w >> 4, h = w & 15;
    int o = otile * 16 + (lane & 15);
    int k = ks * 32 + ((lane >> 4) << 3) + (h < 8 ? h : h + 8);
    float v = (k < MSGN) ? W_ih[o * MSGN + k] : W_hh[o * HIDN + (k - MSGN)];
    wcat[idx] = f2bf(v);
  } else if (idx < 442368) {     // nm_W : 8 otiles x 12 ksteps
    long j = idx - 393216;
    int tile = j >> 9, w = j & 511;
    int otile = tile / 12, ks = tile % 12;
    int lane = w >> 4, h = w & 15;
    int o = otile * 16 + (lane & 15);
    int k = ks * 32 + ((lane >> 4) << 3) + (h < 8 ? h : h + 8);
    nmp[j] = f2bf(nm_W[o * KCAT + k]);
  } else if (idx < 458752) {     // enc_W^T : 16 otiles x 2 ksteps
    long j = idx - 442368;
    int tile = j >> 9, w = j & 511;
    int otile = tile >> 1, ks = tile & 1;
    int lane = w >> 4, h = w & 15;
    int o = otile * 16 + (lane & 15);
    int k = ks * 32 + ((lane >> 4) << 3) + (h < 8 ? h : h + 8);
    encp[j] = f2bf(enc_W[o * INF + k]);
  } else if (idx < 491520) {     // dec_W : 4 heads x 16 ksteps (single 16-wide otile)
    long j = idx - 458752;
    int tile = j >> 9, w = j & 511;
    int p = tile >> 4, ks = tile & 15;
    int lane = w >> 4, h = w & 15;
    int o = lane & 15;
    int k = ks * 32 + ((lane >> 4) << 3) + (h < 8 ? h : h + 8);
    decp[j] = f2bf(dec_W[((long)p * OUTC + o) * HIDN + k]);
  } else if (idx < 524288) {     // xa -> bf16 row-major
    long j = idx - 491520;
    xabf[j] = f2bf(xa[j]);
  } else if (idx < 525312) {     // fused bias
    long j = idx - 524288;
    bsum[j] = b_ih[j] + b_hh[j];
  } else if (idx < 656384) {     // cell = 0
    cell[idx - 525312] = 0.0f;
  }
}

// ---------------- encoder: hid = xa @ enc_W^T + enc_b  (512x256x64, WMMA) ----------------
__global__ void __launch_bounds__(256) encoder(
    const unsigned short* __restrict__ xabf, const unsigned short* __restrict__ encp,
    const float* __restrict__ enc_b, float* __restrict__ hid) {
  int tid = threadIdx.x, lane = tid & 31;
  int w = blockIdx.x * 8 + (tid >> 5);
  int mtile = w >> 4, ntile = w & 15;
  const v16bf* bp = (const v16bf*)encp;
  v16bf a; v8f c = {};
  int row = mtile * 16 + (lane & 15);
  for (int ks = 0; ks < 2; ks++) {
    int kb = ks * 32 + ((lane >> 4) << 3);
#pragma unroll
    for (int h = 0; h < 16; h++) a[h] = bfbits(xabf[row * INF + kb + (h < 8 ? h : h + 8)]);
    v16bf b = bp[(ntile * 2 + ks) * 32 + lane];
    c = __builtin_amdgcn_wmma_f32_16x16x32_bf16(false, a, false, b, (short)0, c, false, false);
  }
  int col = ntile * 16 + (lane & 15);
  int rbase = mtile * 16 + ((lane >> 4) << 3);
  float bb = enc_b[col];
#pragma unroll
  for (int j = 0; j < 8; j++) hid[(rbase + j) * HIDN + col] = c[j] + bb;
}

// ---------------- persistent sequential LSTM queue scan (1 workgroup) ----------------
__global__ void __launch_bounds__(1024) lstm_scan(
    const int* __restrict__ hdr, const int* __restrict__ degs, const int* __restrict__ nbrs,
    int* __restrict__ q_nodes, float* __restrict__ q_msgs,
    float* __restrict__ hid, float* __restrict__ cell,
    const unsigned short* __restrict__ wcat, const unsigned short* __restrict__ nmp,
    const float* __restrict__ bsum, const float* __restrict__ nm_b) {
  // nm_W fragments resident in LDS for all 5120 steps (96 KB of the WGP's 320 KB)
  __shared__ __align__(64) unsigned short nm_lds[8 * 12 * 512];
  __shared__ __align__(64) unsigned short x_bf[KCAT];   // [msg | h_prev] bf16
  __shared__ __align__(64) unsigned short y_bf[KCAT];   // [h_new | msg] bf16
  __shared__ __align__(64) float cprev[HIDN];
  __shared__ float gatesS[GATES];
  __shared__ float nmsg[MSGN];
  __shared__ int s_node, s_t;
  int tid = threadIdx.x;
  int lane = tid & 31, wv = tid >> 5;
  if (tid == 0) s_t = hdr[0];

  // ---- TDM: DMA the 96 KB nm_W fragment buffer global -> LDS once ----
  if (wv == 0) {
    unsigned long ga = (unsigned long)(size_t)nmp;
    v4u g0;
    g0[0] = 1u;                                   // count=1, user descriptor
    g0[1] = (unsigned)(size_t)(&nm_lds[0]);       // lds_addr
    g0[2] = (unsigned)ga;                         // global_addr[31:0]
    g0[3] = ((unsigned)(ga >> 32) & 0x01FFFFFFu) | (2u << 30);  // addr[56:32] | type=2
    v8u g1;
    g1[0] = (3u << 16);                           // data_size = 8 bytes
    g1[1] = (12288u & 0xFFFFu) << 16;             // tensor_dim0[15:0] (12288 x 8B = 96 KB)
    g1[2] = (12288u >> 16) | (1u << 16);          // tensor_dim0[31:16] | tensor_dim1[15:0]=1
    g1[3] = (12288u << 16);                       // tile_dim0 = 12288
    g1[4] = 0u;                                   // tile_dim1 = tile_dim2 = 0 (1-D tile)
    g1[5] = 12288u;                               // tensor_dim0_stride
    g1[6] = 0u;
    g1[7] = 0u;
    asm volatile("tensor_load_to_lds %0, %1" :: "s"(g0), "s"(g1) : "memory");
    __builtin_amdgcn_s_wait_tensorcnt(0);
  }
  __syncthreads();

  const v16bf* wcatv = (const v16bf*)wcat;
  const v16bf* nmlv  = (const v16bf*)nm_lds;

  for (int i = 0; i < MAX_STEPS; i++) {
    if (i >= s_t) break;   // queue empty -> no further state change possible
    if (tid == 0) s_node = q_nodes[i];
    __syncthreads();
    int node = s_node;
    // stage msg / h_prev (bf16 convert) ; c_prev via async global->LDS DMA
    if (tid < MSGN) {
      unsigned short mb = f2bf(q_msgs[(long)i * MSGN + tid]);
      x_bf[tid] = mb;
      y_bf[HIDN + tid] = mb;          // msg half of concat for new_msg
    } else if (tid < MSGN + HIDN) {
      int j = tid - MSGN;
      x_bf[MSGN + j] = f2bf(hid[node * HIDN + j]);
    } else if (wv == 12) {
      // 1 KB cell row: 2 x (32 lanes x 16B) async copies straight into LDS
      unsigned l0 = (unsigned)(size_t)(&cprev[0]) + (unsigned)lane * 16u;
      unsigned long g0 = (unsigned long)(size_t)(cell + (long)node * HIDN) + (unsigned long)lane * 16u;
      unsigned l1 = l0 + 512u;
      unsigned long g1 = g0 + 512u;
      asm volatile("global_load_async_to_lds_b128 %0, %1, off" :: "v"(l0), "v"(g0) : "memory");
      asm volatile("global_load_async_to_lds_b128 %0, %1, off" :: "v"(l1), "v"(g1) : "memory");
      asm volatile("s_wait_asynccnt 0x0" ::: "memory");
    }
    __syncthreads();
    // gates = Wcat^T stacked matvec via WMMA: 64 out-tiles, 2 per wave (B from L2)
    {
      int tile0 = wv * 2, tile1 = tile0 + 1;
      v16bf a; v8f c0 = {}, c1 = {};
      for (int ks = 0; ks < 12; ks++) {
        if ((lane & 15) == 0) {        // only A row 0 carries the vector
          int base = ks * 32 + ((lane >> 4) << 3);
#pragma unroll
          for (int h = 0; h < 16; h++) a[h] = bfbits(x_bf[base + (h < 8 ? h : h + 8)]);
        } else {
#pragma unroll
          for (int h = 0; h < 16; h++) a[h] = bfbits(0);
        }
        v16bf b0 = wcatv[(tile0 * 12 + ks) * 32 + lane];
        v16bf b1 = wcatv[(tile1 * 12 + ks) * 32 + lane];
        if (ks < 11) {
          __builtin_prefetch(&wcatv[(tile0 * 12 + ks + 1) * 32 + lane], 0, 3);
          __builtin_prefetch(&wcatv[(tile1 * 12 + ks + 1) * 32 + lane], 0, 3);
        }
        c0 = __builtin_amdgcn_wmma_f32_16x16x32_bf16(false, a, false, b0, (short)0, c0, false, false);
        c1 = __builtin_amdgcn_wmma_f32_16x16x32_bf16(false, a, false, b1, (short)0, c1, false, false);
      }
      if (lane < 16) {                 // D row M=0 lives in c[0], lanes 0..15
        gatesS[tile0 * 16 + lane] = c0[0] + bsum[tile0 * 16 + lane];
        gatesS[tile1 * 16 + lane] = c1[0] + bsum[tile1 * 16 + lane];
      }
    }
    __syncthreads();
    // LSTM cell elementwise (i,f,g,o)
    if (tid < HIDN) {
      float gi = gatesS[tid], gf = gatesS[HIDN + tid];
      float gg = gatesS[2 * HIDN + tid], go = gatesS[3 * HIDN + tid];
      float si = 1.0f / (1.0f + expf(-gi));
      float sf = 1.0f / (1.0f + expf(-gf));
      float so = 1.0f / (1.0f + expf(-go));
      float cn = sf * cprev[tid] + si * tanhf(gg);
      float hn = so * tanhf(cn);
      cell[node * HIDN + tid] = cn;
      hid[node * HIDN + tid] = hn;
      y_bf[tid] = f2bf(hn);
    }
    __syncthreads();
    // new_msg = nm_W @ [h_new ; msg] + nm_b : 8 out-tiles on waves 0..7, B from LDS
    if (wv < 8) {
      v16bf a; v8f c = {};
      for (int ks = 0; ks < 12; ks++) {
        if ((lane & 15) == 0) {
          int base = ks * 32 + ((lane >> 4) << 3);
#pragma unroll
          for (int h = 0; h < 16; h++) a[h] = bfbits(y_bf[base + (h < 8 ? h : h + 8)]);
        } else {
#pragma unroll
          for (int h = 0; h < 16; h++) a[h] = bfbits(0);
        }
        v16bf b = nmlv[(wv * 12 + ks) * 32 + lane];
        c = __builtin_amdgcn_wmma_f32_16x16x32_bf16(false, a, false, b, (short)0, c, false, false);
      }
      if (lane < 16) nmsg[wv * 16 + lane] = c[0] + nm_b[wv * 16 + lane];
    }
    __syncthreads();
    // enqueue exactly the committed degs[node] slots (reads never pass MAX_STEPS)
    int t = s_t, dg = degs[node];
    for (int j = tid; j < dg * MSGN; j += 1024) {
      int q = j >> 7, c = j & 127;
      int slot = t + q;
      if (slot < QCAP) q_msgs[(long)slot * MSGN + c] = nmsg[c];
    }
    if (tid < dg) {
      int slot = t + tid;
      if (slot < QCAP) q_nodes[slot] = nbrs[node * MAXD + tid];
    }
    __syncthreads();
    if (tid == 0) s_t = t + dg;
    __syncthreads();
  }
}

// ---------------- decoder: logits + log_softmax (WMMA + shfl reductions) ----------------
__global__ void __launch_bounds__(256) decoder(
    const float* __restrict__ hid, const unsigned short* __restrict__ decp,
    const float* __restrict__ dec_b, float* __restrict__ out) {
  int tid = threadIdx.x, lane = tid & 31;
  int w = blockIdx.x * 8 + (tid >> 5);
  int mtile = w >> 2, p = w & 3;
  const v16bf* bp = (const v16bf*)decp;
  v16bf a; v8f c = {};
  int row = mtile * 16 + (lane & 15);
  for (int ks = 0; ks < 16; ks++) {
    int kb = ks * 32 + ((lane >> 4) << 3);
#pragma unroll
    for (int h = 0; h < 16; h++)
      a[h] = bfbits(f2bf(hid[row * HIDN + kb + (h < 8 ? h : h + 8)]));
    v16bf b = bp[(p * 16 + ks) * 32 + lane];
    c = __builtin_amdgcn_wmma_f32_16x16x32_bf16(false, a, false, b, (short)0, c, false, false);
  }
  int col = lane & 15;
  float bb = dec_b[p * OUTC + col];
  int rbase = mtile * 16 + ((lane >> 4) << 3);
#pragma unroll
  for (int j = 0; j < 8; j++) {
    float v = c[j] + bb;
    float mx = v;
    for (int m = 8; m >= 1; m >>= 1) mx = fmaxf(mx, __shfl_xor(mx, m, 32));
    float ex = expf(v - mx), sum = ex;
    for (int m = 8; m >= 1; m >>= 1) sum += __shfl_xor(sum, m, 32);
    out[((long)p * NND + rbase + j) * OUTC + col] = v - mx - logf(sum);
  }
}

extern "C" void kernel_launch(void* const* d_in, const int* in_sizes, int n_in,
                              void* d_out, int out_size, void* d_ws, size_t ws_size,
                              hipStream_t stream) {
  const float* xa      = (const float*)d_in[0];
  const float* fmsg    = (const float*)d_in[1];
  const int*   eidx    = (const int*)d_in[2];
  const int*   starts  = (const int*)d_in[3];
  const float* enc_W   = (const float*)d_in[4];
  const float* enc_b   = (const float*)d_in[5];
  const float* W_ih    = (const float*)d_in[6];
  const float* W_hh    = (const float*)d_in[7];
  const float* b_ih    = (const float*)d_in[8];
  const float* b_hh    = (const float*)d_in[9];
  const float* nm_W    = (const float*)d_in[10];
  const float* nm_b    = (const float*)d_in[11];
  const float* dec_W   = (const float*)d_in[12];
  const float* dec_b   = (const float*)d_in[13];

  char* ws = (char*)d_ws;
  int*   hdr   = (int*)(ws + OFF_HDR);
  int*   degs  = (int*)(ws + OFF_DEGS);
  int*   nbrs  = (int*)(ws + OFF_NBRS);
  int*   qn    = (int*)(ws + OFF_QN);
  float* qm    = (float*)(ws + OFF_QM);
  float* hid   = (float*)(ws + OFF_HID);
  float* cell  = (float*)(ws + OFF_CELL);
  unsigned short* wcat = (unsigned short*)(ws + OFF_WCAT);
  unsigned short* nmp  = (unsigned short*)(ws + OFF_NMP);
  unsigned short* encp = (unsigned short*)(ws + OFF_ENCP);
  unsigned short* decp = (unsigned short*)(ws + OFF_DECP);
  unsigned short* xabf = (unsigned short*)(ws + OFF_XABF);
  float* bsum  = (float*)(ws + OFF_BSUM);

  prep_graph<<<1, 1024, 0, stream>>>(eidx, starts, fmsg, hdr, degs, nbrs, qn, qm);
  pack_weights<<<2564, 256, 0, stream>>>(W_ih, W_hh, nm_W, enc_W, dec_W, xa,
                                         b_ih, b_hh, wcat, nmp, encp, decp,
                                         xabf, bsum, cell);
  encoder<<<64, 256, 0, stream>>>(xabf, encp, enc_b, hid);
  lstm_scan<<<1, 1024, 0, stream>>>(hdr, degs, nbrs, qn, qm, hid, cell,
                                    wcat, nmp, bsum, nm_b);
  decoder<<<16, 256, 0, stream>>>(hid, decp, dec_b, (float*)d_out);
}
